// RSMLayer_76622216561438
// MI455X (gfx1250) — compile-verified, matrix operand order checked by reading, not run on hardware
//
#include <hip/hip_runtime.h>
#include <cstdint>
#include <cstddef>

// ---------------- problem constants (from reference) ----------------
#define MG      200     // groups (m)
#define NC      6       // cells per group (n)
#define KTOP    25      // winning columns
#define TCELLS  1200    // total cells = MG*NC
#define DIN     784
#define DOUT    784
#define BSZ     2048
#define SEQ     16

// padded K (multiples of 32 for v_wmma_f32_16x16x32_f16)
#define LD_XA   800     // 784 -> 800
#define LD_XB   1216    // 1200 -> 1216
#define LD_G    224     // 200 -> 224

#define LDS_STRIDE 40   // halves per staged row (80B: 16B-aligned, bank-spread)

typedef _Float16 v16h __attribute__((ext_vector_type(16)));
typedef _Float16 v8h  __attribute__((ext_vector_type(8)));
typedef float    v8f  __attribute__((ext_vector_type(8)));

// order-preserving float<->uint mapping for atomic min on any-sign floats
__device__ __forceinline__ unsigned f2o(float f) {
  unsigned u = __float_as_uint(f);
  return (u & 0x80000000u) ? ~u : (u | 0x80000000u);
}
__device__ __forceinline__ float o2f(unsigned m) {
  return __uint_as_float((m & 0x80000000u) ? (m & 0x7FFFFFFFu) : ~m);
}

// assemble a 16-half WMMA fragment from two 16B LDS chunks (+0, +16 halves)
__device__ __forceinline__ v16h ldfrag(const _Float16* p) {
  v8h lo = *(const v8h*)p;
  v8h hi = *(const v8h*)(p + 16);
  return __builtin_shufflevector(lo, hi, 0,1,2,3,4,5,6,7,8,9,10,11,12,13,14,15);
}

// CDNA5 async global->LDS staging (ASYNCcnt-tracked, in-order completion)
__device__ __forceinline__ void async_b128(unsigned lds_off, const _Float16* gaddr) {
  asm volatile("global_load_async_to_lds_b128 %0, %1, off"
               :: "v"(lds_off), "v"(gaddr) : "memory");
}
__device__ __forceinline__ void wait_async_le2() {
  asm volatile("s_wait_asynccnt 0x2" ::: "memory");
}
__device__ __forceinline__ void wait_async_0() {
  asm volatile("s_wait_asynccnt 0x0" ::: "memory");
}

// -------- f32 -> f16 row-major convert with zero-padded leading dim --------
__global__ void cvt_pad_f16(const float* __restrict__ src, _Float16* __restrict__ dst,
                            int rows, int cols, int ldd) {
  size_t n = (size_t)rows * (size_t)ldd;
  for (size_t i = (size_t)blockIdx.x * blockDim.x + threadIdx.x; i < n;
       i += (size_t)gridDim.x * blockDim.x) {
    int c = (int)(i % (size_t)ldd);
    int r = (int)(i / (size_t)ldd);
    dst[i] = (c < cols) ? (_Float16)src[(size_t)r * cols + c] : (_Float16)0.0f;
  }
}

// ---------------------------------------------------------------------------
// C[r,c] = sum_k A[r,k]*W[c,k] + bias[c] (+ za[r, c/NC] if za)        (f32 out)
// 64x64 macro-tile per 256-thread block; double-buffered async global->LDS
// staging of 64x32 A/B K-slices; 8 waves, each wave = 16x32 slab (2 WMMA accs).
// Optional fused global-min reduction. rows%64==0, K%32==0 (zero padded).
// ---------------------------------------------------------------------------
__global__ __launch_bounds__(256)
void wmma_gemm_f16(const _Float16* __restrict__ A, int lda,
                   const _Float16* __restrict__ W, int ldw,
                   const float* __restrict__ bias,
                   const float* __restrict__ za,
                   float* __restrict__ C, int ldc,
                   int cols, int K, int macroCols,
                   unsigned* __restrict__ minOut,
                   unsigned* __restrict__ minReset) {
  if (minReset && blockIdx.x == 0 && threadIdx.x == 0) *minReset = 0xFFFFFFFFu;

  __shared__ _Float16 As[2][64 * LDS_STRIDE];
  __shared__ _Float16 Bs[2][64 * LDS_STRIDE];

  const int rowT = blockIdx.x / macroCols;
  const int colT = blockIdx.x - rowT * macroCols;
  const int row0 = rowT << 6;
  const int col0 = colT << 6;

  // ---- staging addresses (one 16B A-chunk + one 16B B-chunk per thread) ----
  const int sRow = threadIdx.x >> 2;            // 0..63
  const int sChk = (threadIdx.x & 3) << 3;      // 0,8,16,24 halves
  const _Float16* gA = A + (size_t)(row0 + sRow) * lda + sChk;
  int bcol = col0 + sRow; if (bcol > cols - 1) bcol = cols - 1;   // clamp OOB reads
  const _Float16* gB = W + (size_t)bcol * ldw + sChk;
  const unsigned ldsOff = (unsigned)((sRow * LDS_STRIDE + sChk) * 2);
  const unsigned ldsA[2] = { (unsigned)(size_t)&As[0][0] + ldsOff,
                             (unsigned)(size_t)&As[1][0] + ldsOff };
  const unsigned ldsB[2] = { (unsigned)(size_t)&Bs[0][0] + ldsOff,
                             (unsigned)(size_t)&Bs[1][0] + ldsOff };

  // ---- per-wave compute mapping: wave w -> rows 16*(w>>1), cols 32*(w&1) ----
  const int lane = threadIdx.x & 31;
  const int wave = threadIdx.x >> 5;
  const int wr = wave >> 1;                     // 0..3
  const int wc = wave & 1;                      // 0..1
  const int mi = lane & 15;
  const int hi = lane >> 4;
  const int aoff  = (16 * wr + mi) * LDS_STRIDE + 8 * hi;
  const int boff0 = (32 * wc + mi) * LDS_STRIDE + 8 * hi;
  const int boff1 = boff0 + 16 * LDS_STRIDE;

  v8f acc0 = {0.f,0.f,0.f,0.f,0.f,0.f,0.f,0.f};
  v8f acc1 = {0.f,0.f,0.f,0.f,0.f,0.f,0.f,0.f};

  const int nk = K >> 5;
  // prologue: stage K-slice 0 into buffer 0 (2 async ops per wave)
  async_b128(ldsA[0], gA);
  async_b128(ldsB[0], gB);
  int buf = 0;
  for (int kc = 0; kc < nk; ++kc) {
    if (kc + 1 < nk) {                          // stage next slice into buf^1
      const int kn = (kc + 1) << 5;
      async_b128(ldsA[buf ^ 1], gA + kn);
      async_b128(ldsB[buf ^ 1], gB + kn);
      wait_async_le2();                         // only the 2 newer ops may remain
    } else {
      wait_async_0();
    }
    __syncthreads();                            // slice `buf` visible to all waves

    v16h a  = ldfrag(&As[buf][aoff]);
    v16h b0 = ldfrag(&Bs[buf][boff0]);
    v16h b1 = ldfrag(&Bs[buf][boff1]);
    acc0 = __builtin_amdgcn_wmma_f32_16x16x32_f16(false, a, false, b0,
                                                  (short)0, acc0, false, false);
    acc1 = __builtin_amdgcn_wmma_f32_16x16x32_f16(false, a, false, b1,
                                                  (short)0, acc1, false, false);
    __syncthreads();                            // done reading before next overwrite
    buf ^= 1;
  }

  // ---- epilogue: bias, optional z_a repeat-add, store, fused min ----
  float lmin = 3.4e38f;
  const int crowBase = row0 + 16 * wr + 8 * hi; // C layout: M = r + 8*hi
#pragma unroll
  for (int ct = 0; ct < 2; ++ct) {
    const v8f acc = ct ? acc1 : acc0;
    const int ccol = col0 + 32 * wc + 16 * ct + mi;
    if (ccol < cols) {
      const float bv = bias ? bias[ccol] : 0.0f;
      const int grp = ccol / NC;
#pragma unroll
      for (int r = 0; r < 8; ++r) {
        const int crow = crowBase + r;
        float v = acc[r] + bv;
        if (za) v += za[(size_t)crow * MG + grp];
        C[(size_t)crow * ldc + ccol] = v;
        lmin = fminf(lmin, v);
      }
    }
  }
  if (minOut) {
    unsigned um = f2o(lmin);
#pragma unroll
    for (int off = 16; off > 0; off >>= 1) {
      unsigned other = (unsigned)__shfl_xor((int)um, off, 32);
      um = (other < um) ? other : um;
    }
    if (lane == 0) atomicMin(minOut, um);
  }
}

// ---------------------------------------------------------------------------
// Per batch row: pi, group-max lam, top-1-per-group mask, top-25 column mask
// (count-greater == "lam >= 25th largest" incl. ties), y = tanh(mask*sigma),
// group max g, in-place phi/psi update, x_b_n emitted f32 (output) + f16 (next GEMM).
// ---------------------------------------------------------------------------
__global__ __launch_bounds__(256)
void mask_update(const float* __restrict__ sigma, const unsigned* __restrict__ minv,
                 float* __restrict__ phi, float* __restrict__ psi,
                 float* __restrict__ xbs_t,
                 _Float16* __restrict__ xbh, _Float16* __restrict__ gh) {
  const int b = blockIdx.x;
  const float mn = o2f(*minv);

  __shared__ float s_pi[TCELLS];
  __shared__ float s_sig[TCELLS];
  __shared__ float s_lam[MG];
  __shared__ float s_flag[MG];

  const float* srow   = sigma + (size_t)b * TCELLS;
  float*       phirow = phi   + (size_t)b * TCELLS;
  float*       psirow = psi   + (size_t)b * TCELLS;
  float*       xbrow  = xbs_t + (size_t)b * TCELLS;
  _Float16*    xbhrow = xbh   + (size_t)b * LD_XB;
  _Float16*    ghrow  = gh    + (size_t)b * LD_G;

  for (int c = threadIdx.x; c < TCELLS; c += blockDim.x) {
    const float s = srow[c];
    s_sig[c] = s;
    s_pi[c] = (1.0f - phirow[c]) * (s - mn + 1.0f);
  }
  __syncthreads();
  for (int m = threadIdx.x; m < MG; m += blockDim.x) {
    float mx = s_pi[m * NC];
#pragma unroll
    for (int j = 1; j < NC; ++j) mx = fmaxf(mx, s_pi[m * NC + j]);
    s_lam[m] = mx;
  }
  __syncthreads();
  for (int m = threadIdx.x; m < MG; m += blockDim.x) {
    const float v = s_lam[m];
    int cnt = 0;
    for (int j = 0; j < MG; ++j) cnt += (s_lam[j] > v) ? 1 : 0;
    s_flag[m] = (cnt < KTOP) ? 1.0f : 0.0f;      // == (lam >= 25th largest)
  }
  __syncthreads();
  for (int m = threadIdx.x; m < MG; m += blockDim.x) {
    const float fl = s_flag[m];
    const float lm = s_lam[m];
    float gv = -3.4e38f;
#pragma unroll
    for (int j = 0; j < NC; ++j) {
      const int c = m * NC + j;
      const float mk = (s_pi[c] >= lm) ? fl : 0.0f;   // m_pi * m_lam
      const float y = tanhf(mk * s_sig[c]);
      gv = fmaxf(gv, y);
      const float psn = fmaxf(psirow[c] * 0.5f, y);   // EPS = 0.5
      const float phn = fmaxf(phirow[c] * 0.5f, y);   // GAMMA = 0.5
      psirow[c] = psn;
      phirow[c] = phn;
      xbrow[c]  = psn;                                // PRED_GAIN = 1
      xbhrow[c] = (_Float16)psn;
    }
    ghrow[m] = (_Float16)gv;
  }
  for (int c = TCELLS + threadIdx.x; c < LD_XB; c += blockDim.x) xbhrow[c] = (_Float16)0.f;
  for (int m = MG + threadIdx.x; m < LD_G; m += blockDim.x)      ghrow[m]  = (_Float16)0.f;
}

// ---------------------------------------------------------------------------
extern "C" void kernel_launch(void* const* d_in, const int* in_sizes, int n_in,
                              void* d_out, int out_size, void* d_ws, size_t ws_size,
                              hipStream_t stream) {
  (void)in_sizes; (void)n_in; (void)out_size; (void)ws_size;
  const float* x_a  = (const float*)d_in[0];   // [SEQ, BSZ, DIN]
  const float* xb0  = (const float*)d_in[1];   // [BSZ, TCELLS]
  const float* phi0 = (const float*)d_in[2];
  const float* psi0 = (const float*)d_in[3];
  const float* Wa   = (const float*)d_in[4];   // [MG, DIN]
  const float* ba   = (const float*)d_in[5];
  const float* Wb   = (const float*)d_in[6];   // [TCELLS, TCELLS]
  const float* bb   = (const float*)d_in[7];
  const float* Wd   = (const float*)d_in[8];   // [DOUT, MG]
  const float* bd   = (const float*)d_in[9];

  // outputs: output, x_b, phi, psi, x_bs (concatenated)
  float* out   = (float*)d_out;
  float* oPred = out;
  float* oXb   = oPred + (size_t)SEQ * BSZ * DOUT;
  float* oPhi  = oXb  + (size_t)BSZ * TCELLS;
  float* oPsi  = oPhi + (size_t)BSZ * TCELLS;
  float* oXbs  = oPsi + (size_t)BSZ * TCELLS;

  // workspace carve-up (all 16B-aligned by construction, ~24.3 MB total)
  char* w = (char*)d_ws;
  float*    zaBuf = (float*)w;     w += (size_t)BSZ * MG * 4;
  float*    sigma = (float*)w;     w += (size_t)BSZ * TCELLS * 4;
  _Float16* xah   = (_Float16*)w;  w += (size_t)BSZ * LD_XA * 2;
  _Float16* xbh   = (_Float16*)w;  w += (size_t)BSZ * LD_XB * 2;
  _Float16* gh    = (_Float16*)w;  w += (size_t)BSZ * LD_G * 2;
  _Float16* Wah   = (_Float16*)w;  w += (size_t)MG * LD_XA * 2;
  _Float16* Wbh   = (_Float16*)w;  w += (size_t)TCELLS * LD_XB * 2;
  _Float16* Wdh   = (_Float16*)w;  w += (size_t)DOUT * LD_G * 2;
  unsigned* minv  = (unsigned*)w;

  // one-time (per call) f16 staging of weights + initial recurrent state
  cvt_pad_f16<<<512, 256, 0, stream>>>(Wa, Wah, MG, DIN, LD_XA);
  cvt_pad_f16<<<2048, 256, 0, stream>>>(Wb, Wbh, TCELLS, TCELLS, LD_XB);
  cvt_pad_f16<<<512, 256, 0, stream>>>(Wd, Wdh, DOUT, MG, LD_G);
  cvt_pad_f16<<<2048, 256, 0, stream>>>(xb0, xbh, BSZ, TCELLS, LD_XB);
  hipMemcpyAsync(oPhi, phi0, (size_t)BSZ * TCELLS * 4, hipMemcpyDeviceToDevice, stream);
  hipMemcpyAsync(oPsi, psi0, (size_t)BSZ * TCELLS * 4, hipMemcpyDeviceToDevice, stream);

  const int macroRows = BSZ / 64;                      // 32
  const int mcZa = (MG + 63) / 64;                     // 4
  const int mcSg = (TCELLS + 63) / 64;                 // 19
  const int mcDc = (DOUT + 63) / 64;                   // 13

  for (int t = 0; t < SEQ; ++t) {
    cvt_pad_f16<<<2048, 256, 0, stream>>>(x_a + (size_t)t * BSZ * DIN, xah, BSZ, DIN, LD_XA);
    // z_a = x_a @ Wa^T + ba   (also resets the global-min cell for this step)
    wmma_gemm_f16<<<macroRows * mcZa, 256, 0, stream>>>(
        xah, LD_XA, Wah, LD_XA, ba, nullptr, zaBuf, MG,
        MG, LD_XA, mcZa, nullptr, minv);
    // sigma = x_b @ Wb^T + bb + repeat(z_a)  + fused global-min reduction
    wmma_gemm_f16<<<macroRows * mcSg, 256, 0, stream>>>(
        xbh, LD_XB, Wbh, LD_XB, bb, zaBuf, sigma, TCELLS,
        TCELLS, LD_XB, mcSg, minv, nullptr);
    // masks / tanh / group max / state update
    mask_update<<<BSZ, 256, 0, stream>>>(
        sigma, minv, oPhi, oPsi, oXbs + (size_t)t * BSZ * TCELLS, xbh, gh);
    // x_a_pred = g @ Wd^T + bd
    wmma_gemm_f16<<<macroRows * mcDc, 256, 0, stream>>>(
        gh, LD_G, Wdh, LD_G, bd, nullptr, oPred + (size_t)t * BSZ * DOUT, DOUT,
        DOUT, LD_G, mcDc, nullptr, nullptr);
  }
  // final x_b output = last x_bs slice
  hipMemcpyAsync(oXb, oXbs + (size_t)(SEQ - 1) * BSZ * TCELLS,
                 (size_t)BSZ * TCELLS * 4, hipMemcpyDeviceToDevice, stream);
}
